// DiscoEncodeProcessDecode_26869315403890
// MI455X (gfx1250) — compile-verified, matrix operand order checked by reading, not run on hardware
//
#include <hip/hip_runtime.h>
#include <hip/hip_bf16.h>

// ---------------- problem constants (from reference) ----------------
#define Bc   2
#define Hc   128
#define Wc   256
#define MDc  34
#define LDc  32
#define NHc  4
#define HDc  8
#define SHc  64
#define Kc   25
#define Pc   7
#define Rc   3
#define FF1c 512
#define FF2c 256

#define WT   16          // longitude tile per block (kernel 1)
#define WTP  (WT + 2*Rc) // 22 taps wide

typedef __attribute__((ext_vector_type(16))) _Float16 v16h;
typedef __attribute__((ext_vector_type(8)))  _Float16 v8h;
typedef __attribute__((ext_vector_type(8)))  float    v8f;

union FragU { v16h v; v8h h[2]; };

// Load a 16x32 f16 A/B fragment from LDS (row-major [rows][stride]),
// per CDNA5 ISA layout: lanes 0-15 -> M rows, K 0-7 & 16-23;
// lanes 16-31 -> same M rows, K 8-15 & 24-31. Two ds_load_b128 per lane.
__device__ inline v16h ld_frag(const _Float16* base, int stride, int lane, int kk) {
  int m     = lane & 15;
  int khalf = lane >> 4;
  const _Float16* p = base + m * stride + kk + khalf * 8;
  FragU f;
  f.h[0] = *(const v8h*)(p);
  f.h[1] = *(const v8h*)(p + 16);
  return f.v;
}

__device__ inline float gelu_exact(float x) {
  return 0.5f * x * (1.0f + erff(x * 0.70710678118654752f));
}

// =====================================================================
// Kernel 1: DISCO stencil + basis contraction + disco_w + head MLP
// grid: (W/WT, H, B), block: 128 threads (4 waves)
// min 4 blocks/EU -> cap VGPRs <= 256/lane (no vgpr-MSB addressing)
// =====================================================================
__global__ __launch_bounds__(128, 4)
void disco_head_kernel(const float* __restrict__ x,
                       const float* __restrict__ psi,
                       const float* __restrict__ dw,
                       const float* __restrict__ db,
                       const float* __restrict__ hw1,
                       const float* __restrict__ hb1,
                       const float* __restrict__ hw2,
                       const float* __restrict__ hb2,
                       float* __restrict__ xl_out) {
  __shared__ __align__(16) _Float16 xt[Pc][WTP][LDc];      // input taps (f16)
  __shared__ __align__(16) _Float16 psiS[Pc * Pc][Kc];     // psi[p][k] for this latitude
  __shared__ __align__(16) _Float16 wB[SHc][32];           // disco_w^T, K padded 25->32
  __shared__ __align__(16) float    dbS[SHc];
  __shared__ __align__(16) _Float16 hw1S[32][SHc];         // per-head: [out col][k]
  __shared__ __align__(16) float    hb1S[32];
  __shared__ __align__(16) float    hw2S[32];
  __shared__               float    hb2S;
  __shared__ __align__(16) _Float16 basisS[128][32];       // rows = (lon,chan)
  __shared__ __align__(16) _Float16 dS[128][SHc];
  __shared__ __align__(16) float    hS[128][32];

  const int tid  = threadIdx.x;
  const int lane = tid & 31;
  const int wv   = tid >> 5;
  const int w0 = blockIdx.x * WT;
  const int h  = blockIdx.y;
  const int b  = blockIdx.z;

  // ---- stage input tile (lat clamp, lon wrap), psi row, disco_w ----
  for (int idx = tid; idx < Pc * WTP * LDc; idx += 128) {
    int dlat = idx / (WTP * LDc);
    int rem  = idx - dlat * (WTP * LDc);
    int j    = rem / LDc;
    int n    = rem - j * LDc;
    int hr = h - Rc + dlat;  hr = hr < 0 ? 0 : (hr > Hc - 1 ? Hc - 1 : hr);
    int wg = (w0 - Rc + j + Wc) % Wc;
    xt[dlat][j][n] = (_Float16)x[(((size_t)b * Hc + hr) * Wc + wg) * MDc + n];
  }
  for (int idx = tid; idx < Pc * Pc * Kc; idx += 128) {
    int p = idx / Kc;
    int k = idx - p * Kc;
    psiS[p][k] = (_Float16)psi[((size_t)k * Hc + h) * (Pc * Pc) + p];
  }
  for (int idx = tid; idx < SHc * 32; idx += 128) {
    int f = idx >> 5, k = idx & 31;
    wB[f][k] = (k < Kc) ? (_Float16)dw[f * Kc + k] : (_Float16)0.0f;
  }
  for (int idx = tid; idx < SHc; idx += 128) dbS[idx] = db[idx];
  __syncthreads();

  // ---- loop over the 4 heads: 128 rows = 16 lon x 8 channels ----
  for (int g = 0; g < NHc; ++g) {
    __syncthreads();  // previous iteration fully done before reloading weights
    for (int idx = tid; idx < 32 * SHc; idx += 128) {
      int o = idx >> 6, s = idx & 63;
      hw1S[o][s] = (_Float16)hw1[((size_t)(g * SHc + s)) * 32 + o];
    }
    if (tid < 32) { hb1S[tid] = hb1[g * 32 + tid]; hw2S[tid] = hw2[g * 32 + tid]; }
    if (tid == 0) hb2S = hb2[g];

    // ---- basis: 25-vector per row via VALU FMAs against LDS taps ----
    // dlat rolled to keep live state small; K=25 fan-out unrolled for
    // dual-issue alongside ds traffic.
    {
      int row = tid;              // 0..127
      int wl  = row >> 3;         // longitude in tile
      int cl  = row & 7;          // channel within head
      int n   = g * HDc + cl;
      float bas[Kc];
      #pragma unroll
      for (int k = 0; k < Kc; ++k) bas[k] = 0.0f;
      #pragma unroll 1
      for (int dlat = 0; dlat < Pc; ++dlat) {
        #pragma unroll 1
        for (int dlon = 0; dlon < Pc; ++dlon) {
          int p = dlat * Pc + dlon;
          float tap = (float)xt[dlat][wl + dlon][n];
          #pragma unroll
          for (int k = 0; k < Kc; ++k)
            bas[k] = fmaf((float)psiS[p][k], tap, bas[k]);
        }
      }
      #pragma unroll
      for (int k = 0; k < 32; ++k)
        basisS[row][k] = (k < Kc) ? (_Float16)bas[k] : (_Float16)0.0f;
    }
    __syncthreads();

    // ---- GEMM1: [128 x 32] @ [32 x 64] -> dS (disco_w + bias) ----
    #pragma unroll 2
    for (int i = 0; i < 8; ++i) {
      int tile = wv * 8 + i;
      int mt = tile >> 2, nt = tile & 3;
      v16h a = ld_frag(&basisS[mt * 16][0], 32, lane, 0);
      v16h w = ld_frag(&wB[nt * 16][0],     32, lane, 0);
      v8f  c = {};
      c = __builtin_amdgcn_wmma_f32_16x16x32_f16(false, a, false, w,
                                                 (short)0, c, false, false);
      int colL = lane & 15, rh = lane >> 4;
      #pragma unroll
      for (int r = 0; r < 8; ++r) {
        int row = mt * 16 + r + 8 * rh;
        int col = nt * 16 + colL;
        dS[row][col] = (_Float16)(c[r] + dbS[col]);
      }
    }
    __syncthreads();

    // ---- GEMM2: [128 x 64] @ [64 x 32] -> gelu -> hS ----
    #pragma unroll 2
    for (int i = 0; i < 4; ++i) {
      int tile = wv * 4 + i;
      int mt = tile >> 1, nt = tile & 1;
      v8f c = {};
      #pragma unroll
      for (int ks = 0; ks < 2; ++ks) {
        v16h a = ld_frag(&dS[mt * 16][0],     SHc, lane, ks * 32);
        v16h w = ld_frag(&hw1S[nt * 16][0],   SHc, lane, ks * 32);
        c = __builtin_amdgcn_wmma_f32_16x16x32_f16(false, a, false, w,
                                                   (short)0, c, false, false);
      }
      int colL = lane & 15, rh = lane >> 4;
      #pragma unroll
      for (int r = 0; r < 8; ++r) {
        int row = mt * 16 + r + 8 * rh;
        int col = nt * 16 + colL;
        hS[row][col] = gelu_exact(c[r] + hb1S[col]);
      }
    }
    __syncthreads();

    // ---- hw2 contraction (32-dot) + residual -> workspace ----
    {
      int row = tid;
      int wl  = row >> 3;
      int cl  = row & 7;
      int n   = g * HDc + cl;
      float acc = hb2S;
      #pragma unroll
      for (int o = 0; o < 32; ++o) acc = fmaf(hS[row][o], hw2S[o], acc);
      size_t pix = ((size_t)b * Hc + h) * Wc + (w0 + wl);
      float res = x[pix * MDc + n];
      xl_out[pix * LDc + n] = acc + res;
    }
  }
}

// =====================================================================
// Kernel 2: FF MLP 32->512->256->32 + residual + sin/cos passthrough
// grid: (B*H*W/32), block: 128 threads (4 waves)
// =====================================================================
__global__ __launch_bounds__(128, 4)
void ff_mlp_kernel(const float* __restrict__ xl,
                   const float* __restrict__ x,
                   const float* __restrict__ fw1,
                   const float* __restrict__ fb1,
                   const float* __restrict__ fw2,
                   const float* __restrict__ fb2,
                   const float* __restrict__ fw3,
                   const float* __restrict__ fb3,
                   float* __restrict__ out) {
  __shared__ __align__(16) _Float16 A1[32][32];       // xl tile
  __shared__ __align__(16) _Float16 wS[FF2c][64];     // weight staging (32 KB)
  __shared__ __align__(16) _Float16 A2[32][64];       // gelu(t1) chunk
  __shared__ __align__(16) _Float16 T2[32][FF2c];     // gelu(t2)

  const int tid  = threadIdx.x;
  const int lane = tid & 31;
  const int wv   = tid >> 5;
  const size_t base = (size_t)blockIdx.x * 32;
  _Float16* wFlat = &wS[0][0];

  // stage xl tile (the only channels feeding FF1: fw1 rows 32,33 are zero)
  for (int idx = tid; idx < 32 * 32; idx += 128) {
    int p = idx >> 5, ch = idx & 31;
    A1[p][ch] = (_Float16)xl[(base + p) * LDc + ch];
  }
  __syncthreads();

  // register accumulators for GEMM2: 8 C-tiles per wave (M2 x N16 tiles)
  v8f c2[8];
  #pragma unroll
  for (int i = 0; i < 8; ++i) c2[i] = (v8f){};

  #pragma unroll 1
  for (int kc = 0; kc < FF1c / 64; ++kc) {
    if (kc + 1 < FF1c / 64)
      __builtin_prefetch(&fw2[(size_t)((kc + 1) * 64) * FF2c], 0, 1);

    // fw1 chunk: 64 output cols x 32 k  ([col][k] layout)
    for (int idx = tid; idx < 64 * 32; idx += 128) {
      int col = idx >> 5, k = idx & 31;
      wFlat[col * 32 + k] = (_Float16)fw1[(size_t)k * FF1c + (kc * 64 + col)];
    }
    __syncthreads();

    // GEMM1 chunk: [32x32] @ [32x64] -> gelu -> A2
    #pragma unroll 2
    for (int i2 = 0; i2 < 2; ++i2) {
      int tile = wv * 2 + i2;
      int mt = tile >> 2, nt = tile & 3;
      v16h a = ld_frag(&A1[mt * 16][0],      32, lane, 0);
      v16h w = ld_frag(&wFlat[nt * 16 * 32], 32, lane, 0);
      v8f  c = {};
      c = __builtin_amdgcn_wmma_f32_16x16x32_f16(false, a, false, w,
                                                 (short)0, c, false, false);
      int colL = lane & 15, rh = lane >> 4;
      #pragma unroll
      for (int r = 0; r < 8; ++r) {
        int row = mt * 16 + r + 8 * rh;
        int col = nt * 16 + colL;
        float v = c[r] + fb1[kc * 64 + col];
        A2[row][col] = (_Float16)gelu_exact(v);
      }
    }
    __syncthreads();

    // fw2 chunk: 256 output cols x 64 k
    for (int idx = tid; idx < FF2c * 64; idx += 128) {
      int col = idx >> 6, k = idx & 63;
      wFlat[col * 64 + k] = (_Float16)fw2[(size_t)(kc * 64 + k) * FF2c + col];
    }
    __syncthreads();

    // GEMM2 accumulate: [32x64] @ [64x256] into register C-tiles
    #pragma unroll 2
    for (int i = 0; i < 8; ++i) {
      int tile = wv * 8 + i;
      int mt = tile >> 4, nt = tile & 15;
      #pragma unroll
      for (int ks = 0; ks < 2; ++ks) {
        v16h a = ld_frag(&A2[mt * 16][0],      64, lane, ks * 32);
        v16h w = ld_frag(&wFlat[nt * 16 * 64], 64, lane, ks * 32);
        c2[i] = __builtin_amdgcn_wmma_f32_16x16x32_f16(false, a, false, w,
                                                       (short)0, c2[i], false, false);
      }
    }
    __syncthreads();
  }

  // epilogue of GEMM2: bias + gelu -> T2
  {
    int colL = lane & 15, rh = lane >> 4;
    #pragma unroll 2
    for (int i = 0; i < 8; ++i) {
      int tile = wv * 8 + i;
      int mt = tile >> 4, nt = tile & 15;
      #pragma unroll
      for (int r = 0; r < 8; ++r) {
        int row = mt * 16 + r + 8 * rh;
        int col = nt * 16 + colL;
        T2[row][col] = (_Float16)gelu_exact(c2[i][r] + fb2[col]);
      }
    }
  }
  // fw3: 32 output cols x 256 k
  for (int idx = tid; idx < 32 * FF2c; idx += 128) {
    int col = idx >> 8, k = idx & 255;
    wFlat[col * FF2c + k] = (_Float16)fw3[(size_t)k * LDc + col];
  }
  __syncthreads();

  // GEMM3: [32x256] @ [256x32] + bias + residual -> d_out
  {
    int mt = wv >> 1, nt = wv & 1;
    v8f c = {};
    #pragma unroll 2
    for (int ks = 0; ks < 8; ++ks) {
      v16h a = ld_frag(&T2[mt * 16][0],        FF2c, lane, ks * 32);
      v16h w = ld_frag(&wFlat[nt * 16 * FF2c], FF2c, lane, ks * 32);
      c = __builtin_amdgcn_wmma_f32_16x16x32_f16(false, a, false, w,
                                                 (short)0, c, false, false);
    }
    int colL = lane & 15, rh = lane >> 4;
    #pragma unroll
    for (int r = 0; r < 8; ++r) {
      int row = mt * 16 + r + 8 * rh;
      int col = nt * 16 + colL;
      size_t pix = base + row;
      out[pix * MDc + col] = c[r] + fb3[col] + xl[pix * LDc + col];
    }
  }

  // sin/cos passthrough channels 32,33
  for (int idx = tid; idx < 32 * 2; idx += 128) {
    int p = idx >> 1, ch = LDc + (idx & 1);
    out[(base + p) * MDc + ch] = x[(base + p) * MDc + ch];
  }
}

// =====================================================================
extern "C" void kernel_launch(void* const* d_in, const int* in_sizes, int n_in,
                              void* d_out, int out_size, void* d_ws, size_t ws_size,
                              hipStream_t stream) {
  const float* x    = (const float*)d_in[0];
  const float* psi  = (const float*)d_in[1];
  const float* dw   = (const float*)d_in[2];
  const float* db   = (const float*)d_in[3];
  const float* hw1  = (const float*)d_in[4];
  const float* hb1  = (const float*)d_in[5];
  const float* hw2  = (const float*)d_in[6];
  const float* hb2  = (const float*)d_in[7];
  const float* fw1  = (const float*)d_in[8];
  const float* fb1  = (const float*)d_in[9];
  const float* fw2  = (const float*)d_in[10];
  const float* fb2  = (const float*)d_in[11];
  const float* fw3  = (const float*)d_in[12];
  const float* fb3  = (const float*)d_in[13];
  float* out = (float*)d_out;
  float* xl  = (float*)d_ws;   // B*H*W*32 fp32 intermediate (8.4 MB)

  dim3 g1(Wc / WT, Hc, Bc);           // 16 x 128 x 2 blocks
  disco_head_kernel<<<g1, 128, 0, stream>>>(x, psi, dw, db, hw1, hb1, hw2, hb2, xl);

  int nblk2 = (Bc * Hc * Wc) / 32;    // 2048 blocks
  ff_mlp_kernel<<<nblk2, 128, 0, stream>>>(xl, x, fw1, fb1, fw2, fb2, fw3, fb3, out);
}
